// MultiHeadAttention_12515534701302
// MI455X (gfx1250) — compile-verified, matrix operand order checked by reading, not run on hardware
//
#include <hip/hip_runtime.h>

typedef __attribute__((ext_vector_type(16))) _Float16 v16h;
typedef __attribute__((ext_vector_type(8)))  _Float16 v8h;
typedef __attribute__((ext_vector_type(4)))  _Float16 v4h;
typedef __attribute__((ext_vector_type(8)))  float    v8f;

// 128-bit vector type matching the async-to-LDS builtin's pointee type.
typedef int v4i __attribute__((vector_size(16)));
typedef __attribute__((address_space(1))) v4i v4i_g;   // global
typedef __attribute__((address_space(3))) v4i v4i_l;   // LDS

#if __has_builtin(__builtin_amdgcn_global_load_async_to_lds_b128) && \
    __has_builtin(__builtin_amdgcn_s_wait_asynccnt)
#define HAVE_ASYNC_LDS 1
#else
#define HAVE_ASYNC_LDS 0
#endif

#define LOG2E 1.44269504088896340736f

union Frag16 { v16h v; v8h h[2]; };

// Load one 16x32 f16 A-fragment (or B-fragment from the row-major transpose
// of the B operand) per the CDNA5 WMMA VGPR layout:
//   lane L: row = L&15, halves 0..7 hold K = 8*(L>>4)+0..7,
//                       halves 8..15 hold K = 16+8*(L>>4)+0..7
__device__ __forceinline__ v16h load_frag(const _Float16* base, int stride_h) {
  const int lane = threadIdx.x & 31;
  const int r = lane & 15;
  const int hi = lane >> 4;
  const _Float16* p = base + (size_t)r * stride_h + 8 * hi;
  Frag16 f;
  f.h[0] = *(const v8h*)(p);
  f.h[1] = *(const v8h*)(p + 16);
  return f.v;
}

__device__ __forceinline__ v8f wmma_f16(v16h a, v16h b, v8f c) {
  return __builtin_amdgcn_wmma_f32_16x16x32_f16(false, a, false, b, (short)0, c,
                                                false, false);
}

// Copy 64 contiguous bytes (32 halves) global -> LDS. Async on gfx1250
// (tracked by ASYNCcnt), else direct fallback.
__device__ __forceinline__ void copy64B_to_lds(const _Float16* gsrc,
                                               _Float16* ldst) {
#if HAVE_ASYNC_LDS
  v4i_g* g = (v4i_g*)(unsigned long long)gsrc;
  v4i_l* l = (v4i_l*)(unsigned)(unsigned long long)ldst;
  __builtin_amdgcn_global_load_async_to_lds_b128(g, l, 0, 0);
  __builtin_amdgcn_global_load_async_to_lds_b128(g, l, 16, 0);
  __builtin_amdgcn_global_load_async_to_lds_b128(g, l, 32, 0);
  __builtin_amdgcn_global_load_async_to_lds_b128(g, l, 48, 0);
#else
#pragma unroll
  for (int i = 0; i < 4; ++i) ((uint4*)ldst)[i] = ((const uint4*)gsrc)[i];
#endif
}

// Wait for the *current* tile's async loads; if the next tile's 8 loads have
// already been issued, leave them in flight (ASYNC ops complete in order).
__device__ __forceinline__ void wait_tile(bool next_in_flight) {
#if HAVE_ASYNC_LDS
  if (next_in_flight) __builtin_amdgcn_s_wait_asynccnt(8);
  else                __builtin_amdgcn_s_wait_asynccnt(0);
#else
  (void)next_in_flight;
#endif
}

// ---------------------------------------------------------------------------
// Kernel 0: elementwise f32 -> f16 (for x, W_qkv, W_out).
// ---------------------------------------------------------------------------
__global__ __launch_bounds__(256) void cvt_f16_kernel(
    const float* __restrict__ src, _Float16* __restrict__ dst, int n4) {
  const int i = blockIdx.x * 256 + threadIdx.x;
  if (i < n4) {
    const float4 f = ((const float4*)src)[i];
    v4h h;
    h[0] = (_Float16)f.x; h[1] = (_Float16)f.y;
    h[2] = (_Float16)f.z; h[3] = (_Float16)f.w;
    ((v4h*)dst)[i] = h;
  }
}

// ---------------------------------------------------------------------------
// Shared GEMM: out[4096,N] = A[4096,1024](f16) * Bw[N,1024]^T(f16) + bias.
// 128x128 block tile, 8 waves x (64x32), K-tile 64, double-buffered LDS with
// async global->LDS fills. SCALE_Q folds (1/sqrt(hd))*log2(e) into Q columns
// so the attention softmax can run in the log2 domain (raw v_exp_f32).
// ---------------------------------------------------------------------------
template <int N, bool SCALE_Q, typename OutT>
__global__ __launch_bounds__(256) void gemm_f16_kernel(
    const _Float16* __restrict__ A, const _Float16* __restrict__ Bw,
    const float* __restrict__ bias, OutT* __restrict__ out) {
  const int K = 1024;
  __shared__ _Float16 As[2][128][72];
  __shared__ _Float16 Bs[2][128][72];
  const int row0 = blockIdx.x * 128;
  const int col0 = blockIdx.y * 128;
  const int tid  = threadIdx.x;
  const int wave = tid >> 5;
  const int lane = tid & 31;
  const int wm = (wave >> 2) * 64;
  const int wn = (wave & 3) * 32;
  const int lr  = tid >> 1;          // loader row 0..127
  const int lch = (tid & 1) * 32;    // loader column chunk (halves)
  v8f acc[4][2] = {};

  // issue first tile
  copy64B_to_lds(A  + (size_t)(row0 + lr) * K + lch, &As[0][lr][lch]);
  copy64B_to_lds(Bw + (size_t)(col0 + lr) * K + lch, &Bs[0][lr][lch]);

  int cur = 0;
  for (int k0 = 0; k0 < K; k0 += 64) {
    const bool more = (k0 + 64) < K;
    if (more) {
      copy64B_to_lds(A  + (size_t)(row0 + lr) * K + k0 + 64 + lch,
                     &As[cur ^ 1][lr][lch]);
      copy64B_to_lds(Bw + (size_t)(col0 + lr) * K + k0 + 64 + lch,
                     &Bs[cur ^ 1][lr][lch]);
    }
    wait_tile(more);
    __syncthreads();
#pragma unroll
    for (int kk = 0; kk < 64; kk += 32) {
      v16h af[4], bf[2];
#pragma unroll
      for (int i = 0; i < 4; ++i)
        af[i] = load_frag(&As[cur][wm + 16 * i][kk], 72);
#pragma unroll
      for (int j = 0; j < 2; ++j)
        bf[j] = load_frag(&Bs[cur][wn + 16 * j][kk], 72);
#pragma unroll
      for (int i = 0; i < 4; ++i)
#pragma unroll
        for (int j = 0; j < 2; ++j)
          acc[i][j] = wmma_f16(af[i], bf[j], acc[i][j]);
    }
    __syncthreads();
    cur ^= 1;
  }

  const int cn = lane & 15;
  const int chi = lane >> 4;
#pragma unroll
  for (int j = 0; j < 2; ++j) {
    const int col = col0 + wn + 16 * j + cn;
    const float bv = bias[col];
    const float scale =
        (SCALE_Q && ((col % 192) < 64)) ? (0.125f * LOG2E) : 1.0f;
#pragma unroll
    for (int i = 0; i < 4; ++i)
#pragma unroll
      for (int v = 0; v < 8; ++v) {
        const int m = row0 + wm + 16 * i + v + 8 * chi;
        out[(size_t)m * N + col] = (OutT)((acc[i][j][v] + bv) * scale);
      }
  }
}

// ---------------------------------------------------------------------------
// Flash attention, 64-key blocks. Block = 4 waves = 64 query rows of one (b,h).
// qkv row layout: [h*192 + (0:64 q*0.125*log2e | 64:128 k | 128:192 v)].
// Scores are in the log2 domain; softmax uses exp2.
// ---------------------------------------------------------------------------
__global__ __launch_bounds__(128) void flash_attn_kernel(
    const _Float16* __restrict__ qkv, const float* __restrict__ mask,
    _Float16* __restrict__ ctx) {
  const int S = 2048, RS = 3072, D = 1024, HD = 64;
  const int qb = blockIdx.x, h = blockIdx.y, b = blockIdx.z;
  const int tid = threadIdx.x;
  const int wave = tid >> 5;
  const int lane = tid & 31;
  const int cn = lane & 15;
  const int chi = lane >> 4;
  const int q0 = qb * 64 + wave * 16;
  const size_t base = (size_t)b * S * RS;
  const _Float16* Qp = qkv + base + (size_t)q0 * RS + h * 192;
  const _Float16* Kp = qkv + base + h * 192 + 64;
  const _Float16* Vp = qkv + base + h * 192 + 128;

  __shared__ _Float16 Vt[64][72];      // V tile transposed: Vt[d][k], k=0..63
  __shared__ _Float16 Ps[4][16][64];   // per-wave softmax probs (row-major)

  const v16h qf0 = load_frag(Qp, RS);        // rows q0..q0+15, dims 0..31
  const v16h qf1 = load_frag(Qp + 32, RS);   // dims 32..63

  v8f o_acc[4] = {};
  float m_run[8], l_run[8];
#pragma unroll
  for (int v = 0; v < 8; ++v) { m_run[v] = -1e30f; l_run[v] = 0.0f; }

  for (int j = 0; j < S; j += 64) {
    // ---- scores for 4 key chunks of 16 (log2 domain) ----
    v8f s[4];
#pragma unroll
    for (int c = 0; c < 4; ++c) {
      const _Float16* Kc = Kp + (size_t)(j + 16 * c) * RS;
      v8f t = {};
      t = wmma_f16(qf0, load_frag(Kc, RS), t);
      t = wmma_f16(qf1, load_frag(Kc + 32, RS), t);
      s[c] = t;
    }

    // additive mask (broadcast over b,h); scaled by log2e via fma
    const float* mrow = mask + (size_t)(q0 + 8 * chi) * S + j + cn;
#pragma unroll
    for (int c = 0; c < 4; ++c)
#pragma unroll
      for (int v = 0; v < 8; ++v)
        s[c][v] = fmaf(mrow[(size_t)v * S + 16 * c], LOG2E, s[c][v]);

    // ---- online softmax; row m = v + 8*chi lives across 16 lanes ----
#pragma unroll
    for (int v = 0; v < 8; ++v) {
      float mx = fmaxf(fmaxf(s[0][v], s[1][v]), fmaxf(s[2][v], s[3][v]));
      mx = fmaxf(mx, __shfl_xor(mx, 8));
      mx = fmaxf(mx, __shfl_xor(mx, 4));
      mx = fmaxf(mx, __shfl_xor(mx, 2));
      mx = fmaxf(mx, __shfl_xor(mx, 1));
      const float mnew = fmaxf(m_run[v], mx);
      const float corr = exp2f(m_run[v] - mnew);
      m_run[v] = mnew;
      float p[4];
#pragma unroll
      for (int c = 0; c < 4; ++c) p[c] = exp2f(s[c][v] - mnew);
      float rs = (p[0] + p[1]) + (p[2] + p[3]);
      rs += __shfl_xor(rs, 8);
      rs += __shfl_xor(rs, 4);
      rs += __shfl_xor(rs, 2);
      rs += __shfl_xor(rs, 1);
      l_run[v] = l_run[v] * corr + rs;
#pragma unroll
      for (int f = 0; f < 4; ++f) o_acc[f][v] *= corr;
#pragma unroll
      for (int c = 0; c < 4; ++c)
        Ps[wave][v + 8 * chi][16 * c + cn] = (_Float16)p[c];
    }
    __syncthreads();  // prior iteration's P*V done reading Vt

    // ---- cooperative load + transpose of the 64x64 V tile ----
#pragma unroll
    for (int rep = 0; rep < 4; ++rep) {
      const int chunk = tid + rep * 128;   // 512 chunks of 8 halves
      const int kk = chunk >> 3;           // key row 0..63
      const int dc = (chunk & 7) * 8;      // dim chunk
      const v8h val = *(const v8h*)(Vp + (size_t)(j + kk) * RS + dc);
#pragma unroll
      for (int i = 0; i < 8; ++i) Vt[dc + i][kk] = val[i];
    }
    __syncthreads();

    // ---- O += P * V ----
    const v16h pa0 = load_frag(&Ps[wave][0][0], 64);   // keys j..j+31
    const v16h pa1 = load_frag(&Ps[wave][0][32], 64);  // keys j+32..j+63
#pragma unroll
    for (int f = 0; f < 4; ++f) {
      o_acc[f] = wmma_f16(pa0, load_frag(&Vt[f * 16][0], 72), o_acc[f]);
      o_acc[f] = wmma_f16(pa1, load_frag(&Vt[f * 16][32], 72), o_acc[f]);
    }
  }

  // normalize and write context [B,S,H*hd] (f16)
  _Float16* cp = ctx + (size_t)b * S * D + (size_t)q0 * D + h * HD;
#pragma unroll
  for (int f = 0; f < 4; ++f) {
    const int d = f * 16 + cn;
#pragma unroll
    for (int v = 0; v < 8; ++v) {
      const int m = v + 8 * chi;
      cp[(size_t)m * D + d] = (_Float16)(o_acc[f][v] / l_run[v]);
    }
  }
}

extern "C" void kernel_launch(void* const* d_in, const int* in_sizes, int n_in,
                              void* d_out, int out_size, void* d_ws, size_t ws_size,
                              hipStream_t stream) {
  (void)in_sizes; (void)n_in; (void)out_size; (void)ws_size;
  const float* x     = (const float*)d_in[0];
  const float* mask  = (const float*)d_in[1];
  const float* W_qkv = (const float*)d_in[2];
  const float* b_qkv = (const float*)d_in[3];
  const float* W_out = (const float*)d_in[4];
  const float* b_out = (const float*)d_in[5];
  float* out = (float*)d_out;

  // workspace layout (f16 buffers)
  char* ws = (char*)d_ws;
  _Float16* x_h    = (_Float16*)(ws);                        //  8.0 MB
  _Float16* wqkv_h = (_Float16*)(ws + 8388608);              //  6.0 MB
  _Float16* wout_h = (_Float16*)(ws + 8388608 + 6291456);    //  2.0 MB
  _Float16* qkv_h  = (_Float16*)(ws + 16777216);             // 24.0 MB
  _Float16* ctx_h  = (_Float16*)(ws + 16777216 + 25165824);  //  8.0 MB

  cvt_f16_kernel<<<4096, 256, 0, stream>>>(x,     x_h,    1048576);
  cvt_f16_kernel<<<3072, 256, 0, stream>>>(W_qkv, wqkv_h,  786432);
  cvt_f16_kernel<<<1024, 256, 0, stream>>>(W_out, wout_h,  262144);

  gemm_f16_kernel<3072, true, _Float16>
      <<<dim3(32, 24), 256, 0, stream>>>(x_h, wqkv_h, b_qkv, qkv_h);
  flash_attn_kernel<<<dim3(32, 16, 2), 128, 0, stream>>>(qkv_h, mask, ctx_h);
  gemm_f16_kernel<1024, false, float>
      <<<dim3(32, 8), 256, 0, stream>>>(ctx_h, wout_h, b_out, out);
}
